// DipoleInteraction_18794776887568
// MI455X (gfx1250) — matrix operand order repaired and checked by reading, not compiled, not run on hardware
//
#include <hip/hip_runtime.h>

// ---------------------------------------------------------------------------
// PaiNN dipole interaction for MI455X (gfx1250, wave32, WMMA).
//
// Persistent-wave pair path, per 16-pair tile:
//   GEMM1: [16x32(pad K=20)] x [32x16]x4  (C preloaded with b1) -> ssp -> LDS f16
//   GEMM2: [16x64] x [64x16]x4, 2 K-tiles (C preloaded with b2) -> Wij
//   dipole tensor math -> global_atomic_add_f32 scatter into tensor_i (d_ws)
// Weight B-fragments converted to f16 ONCE per wave, reused across tiles.
// Atom path per 16-atom tile:
//   dq_i = sum_d mu*tensor_i (coalesced, staged f16 in LDS)
//   GEMM: [16x64] x [64x16]x4 with Wt (C preloaded with bt) -> ssp -> out
// ---------------------------------------------------------------------------

typedef __attribute__((ext_vector_type(16))) _Float16 v16h;
typedef __attribute__((ext_vector_type(8)))  _Float16 v8h_t;
typedef __attribute__((ext_vector_type(8)))  float    v8f;
typedef __attribute__((ext_vector_type(4)))  float    v4f;

#define FDIM 64
#define NRBF 20
#define LDSP 72            // padded LDS row (halfs) to avoid bank conflicts
#define WPB  8             // waves per block (256 threads)

__device__ __forceinline__ float ssp_f(float x) {
    // shifted softplus: ln(1+e^x) - ln2, numerically stable
    return fmaxf(x, 0.f) + __logf(1.f + __expf(-fabsf(x))) - 0.69314718056f;
}

__device__ __forceinline__ v8f wmma16(v16h a, v16h b, v8f c) {
    // v_wmma_f32_16x16x32_f16: (neg_a, A, neg_b, B, c_mod, C, reuse_a, reuse_b)
    return __builtin_amdgcn_wmma_f32_16x16x32_f16(false, a, false, b,
                                                  (short)0, c, false, false);
}

__device__ __forceinline__ v8f bias_frag(float bias) {
    v8f c;
#pragma unroll
    for (int r = 0; r < 8; ++r) c[r] = bias;
    return c;
}

// B fragment (KxN = 32x16, f16) from row-major f32 weight W[fo][k]:
// out[p, fo] = sum_k in[p,k] * W[fo,k]  =>  B[k][n] = W[f0+n][k]
// Lane layout: col n = lane&15, element i -> k = k0 + (i>>3)*16 + (lane>>4)*8 + (i&7).
__device__ __forceinline__ v16h load_bfrag(const float* __restrict__ W,
                                           int row_stride, int krows,
                                           int f0, int k0, int n, int half) {
    v16h b;
    const int fo = f0 + n;
#pragma unroll
    for (int i = 0; i < 16; ++i) {
        int k = k0 + (i >> 3) * 16 + half * 8 + (i & 7);
        b[i] = (k < krows) ? (_Float16)W[fo * row_stride + k] : (_Float16)0.f;
    }
    return b;
}

// ---------------------------------------------------------------------------
__global__ void zero_kernel(float* __restrict__ p, int count) {
    int i = blockIdx.x * blockDim.x + threadIdx.x;
    if (i < count) p[i] = 0.f;
}

// ---------------------------------------------------------------------------
__global__ __launch_bounds__(32 * WPB)
void pair_kernel(const float* __restrict__ mu_field,
                 const float* __restrict__ f_ij,
                 const float* __restrict__ d_ij,
                 const float* __restrict__ v_ij,
                 const int*   __restrict__ idx_i,
                 const int*   __restrict__ idx_j,
                 const float* __restrict__ rcut_ij,
                 const float* __restrict__ W1,
                 const float* __restrict__ b1,
                 const float* __restrict__ W2,
                 const float* __restrict__ b2,
                 float* __restrict__ tensor_i,
                 int P, int ntiles) {
    __shared__ __align__(16) _Float16 sh[WPB * 16 * LDSP];
    const int lane   = threadIdx.x & 31;
    const int wave   = threadIdx.x >> 5;
    const int wid    = blockIdx.x * WPB + wave;          // global wave id
    const int nwaves = gridDim.x * WPB;
    _Float16* lds = &sh[wave * 16 * LDSP];

    const int n    = lane & 15;
    const int half = lane >> 4;

    // ---- weight fragments + biases: converted ONCE, reused for all tiles ----
    v16h bW1[4], bW2[2][4];
    v8f  cb1[4], cb2[4];
#pragma unroll
    for (int nt = 0; nt < 4; ++nt) {
        bW1[nt]    = load_bfrag(W1, NRBF, NRBF, nt * 16, 0,  n, half);
        bW2[0][nt] = load_bfrag(W2, FDIM, FDIM, nt * 16, 0,  n, half);
        bW2[1][nt] = load_bfrag(W2, FDIM, FDIM, nt * 16, 32, n, half);
        cb1[nt]    = bias_frag(b1[nt * 16 + n]);
        cb2[nt]    = bias_frag(b2[nt * 16 + n]);
    }

    // ---- persistent grid-stride loop over 16-pair tiles (wave-uniform) ----
    for (int tile = wid; tile < ntiles; tile += nwaves) {
        const int p0   = tile * 16;
        const int prow = p0 + n;                 // A row owned by this lane
        const bool rowok = (prow < P);
        const int  pr    = rowok ? prow : 0;

        // prefetch next tile's f_ij row (streaming HBM operand)
        {
            const int pnext = prow + nwaves * 16;
            if (pnext < P) __builtin_prefetch(&f_ij[pnext * NRBF], 0, 3);
        }

        // A fragment from f_ij: 16 pairs x 32K (K padded 20->32).
        // elem i -> k = (i>>3)*16 + half*8 + (i&7); first 8 elems contiguous.
        v16h a1;
        {
            const float* row = &f_ij[pr * NRBF + half * 8];   // k = half*8 ..
            v4f r0 = *(const v4f*)row;                        // k .. k+3
            v4f r1 = *(const v4f*)(row + 4);                  // k+4 .. k+7
#pragma unroll
            for (int i = 0; i < 4; ++i) {
                a1[i]     = rowok ? (_Float16)r0[i] : (_Float16)0.f;
                a1[i + 4] = rowok ? (_Float16)r1[i] : (_Float16)0.f;
            }
#pragma unroll
            for (int i = 8; i < 16; ++i) {        // k = 16 + half*8 + (i-8)
                int k = 16 + half * 8 + (i - 8);
                a1[i] = (rowok && k < NRBF) ? (_Float16)f_ij[pr * NRBF + k]
                                            : (_Float16)0.f;
            }
        }

        // GEMM1 (bias folded into C) + ssp -> f16 staged in LDS [pair][F]
#pragma unroll
        for (int nt = 0; nt < 4; ++nt) {
            v8f c = wmma16(a1, bW1[nt], cb1[nt]);
            const int f = nt * 16 + n;
#pragma unroll
            for (int r = 0; r < 8; ++r) {
                const int m = r + 8 * half;       // C/D row = r + 8*(lane>=16)
                lds[m * LDSP + f] = (_Float16)ssp_f(c[r]);
            }
        }
        // intra-wave cross-lane LDS dependency (stores -> b128 loads)
        asm volatile("s_wait_dscnt 0" ::: "memory");
        __builtin_amdgcn_wave_barrier();

        // A fragments for GEMM2 from LDS (two b128 loads per K-tile)
        v16h a2[2];
#pragma unroll
        for (int kt = 0; kt < 2; ++kt) {
            const _Float16* base = &lds[n * LDSP + kt * 32 + half * 8];
            v8h_t lo = *(const v8h_t*)base;
            v8h_t hi = *(const v8h_t*)(base + 16);
#pragma unroll
            for (int i = 0; i < 8; ++i) { a2[kt][i] = lo[i]; a2[kt][i + 8] = hi[i]; }
        }

        // GEMM2 (K=64 -> 2 K-tiles, bias folded into C)
        v8f acc[4];
#pragma unroll
        for (int nt = 0; nt < 4; ++nt) {
            v8f c = wmma16(a2[0], bW2[0][nt], cb2[nt]);
            acc[nt] = wmma16(a2[1], bW2[1][nt], c);
        }

        // dipole tensor math + L2-resident atomic scatter
#pragma unroll 1
        for (int r = 0; r < 8; ++r) {
            const int  p     = p0 + r + 8 * half;
            const bool valid = (p < P);
            const int  pp    = valid ? p : 0;
            const float rc = rcut_ij[pp];
            const float d  = d_ij[pp];
            const float d2 = d * d;
            const float inv5 = 1.f / (d2 * d2 * d);
            const float v0 = v_ij[pp * 3 + 0];
            const float v1 = v_ij[pp * 3 + 1];
            const float v2 = v_ij[pp * 3 + 2];
            const int   j  = idx_j[pp];
            const int   ia = idx_i[pp];
            const float scale = (valid ? 1.f : 0.f) * rc * inv5;
#pragma unroll
            for (int nt = 0; nt < 4; ++nt) {
                const int   f = nt * 16 + n;
                const float w = acc[nt][r] * scale;        // (h@W2^T+b2)*rcut/d^5
                const int   mb = j * (3 * FDIM) + f;
                const float m0 = mu_field[mb];
                const float m1 = mu_field[mb + FDIM];
                const float m2 = mu_field[mb + 2 * FDIM];
                const float proj = v0 * m0 + v1 * m1 + v2 * m2;
                const float t0 = (m0 * d2 - 3.f * v0 * proj) * w;
                const float t1 = (m1 * d2 - 3.f * v1 * proj) * w;
                const float t2 = (m2 * d2 - 3.f * v2 * proj) * w;
                if (valid) {
                    float* dst = &tensor_i[ia * (3 * FDIM) + f];
                    atomicAdd(dst,            t0);
                    atomicAdd(dst + FDIM,     t1);
                    atomicAdd(dst + 2 * FDIM, t2);
                }
            }
        }
    }
}

// ---------------------------------------------------------------------------
__global__ __launch_bounds__(32 * WPB)
void atom_kernel(const float* __restrict__ mu_field,
                 const float* __restrict__ tensor_i,
                 const float* __restrict__ Wt,
                 const float* __restrict__ bt,
                 float* __restrict__ out,
                 int N, int ntiles) {
    __shared__ __align__(16) _Float16 sh[WPB * 16 * LDSP];
    const int lane   = threadIdx.x & 31;
    const int wave   = threadIdx.x >> 5;
    const int wid    = blockIdx.x * WPB + wave;
    const int nwaves = gridDim.x * WPB;
    _Float16* lds = &sh[wave * 16 * LDSP];

    const int n    = lane & 15;
    const int half = lane >> 4;

    // weight fragments + bias, converted once per wave
    v16h bWt[2][4];
    v8f  cbt[4];
#pragma unroll
    for (int nt = 0; nt < 4; ++nt) {
        bWt[0][nt] = load_bfrag(Wt, FDIM, FDIM, nt * 16, 0,  n, half);
        bWt[1][nt] = load_bfrag(Wt, FDIM, FDIM, nt * 16, 32, n, half);
        cbt[nt]    = bias_frag(bt[nt * 16 + n]);
    }

    for (int tile = wid; tile < ntiles; tile += nwaves) {
        const int a0 = tile * 16;

        // dq_i = sum_d mu*tensor_i, coalesced, staged f16 in LDS [16 atoms][F]
#pragma unroll
        for (int it = 0; it < 32; ++it) {
            const int idx = it * 32 + lane;      // 16 atoms x 64 feats
            const int al  = idx >> 6;
            const int f   = idx & 63;
            const int a   = a0 + al;
            float v = 0.f;
            if (a < N) {
                const int base = a * (3 * FDIM) + f;
                v = mu_field[base]            * tensor_i[base]
                  + mu_field[base + FDIM]     * tensor_i[base + FDIM]
                  + mu_field[base + 2 * FDIM] * tensor_i[base + 2 * FDIM];
            }
            lds[al * LDSP + f] = (_Float16)v;
        }
        asm volatile("s_wait_dscnt 0" ::: "memory");
        __builtin_amdgcn_wave_barrier();

        // A fragments from LDS
        v16h a2[2];
#pragma unroll
        for (int kt = 0; kt < 2; ++kt) {
            const _Float16* base = &lds[n * LDSP + kt * 32 + half * 8];
            v8h_t lo = *(const v8h_t*)base;
            v8h_t hi = *(const v8h_t*)(base + 16);
#pragma unroll
            for (int i = 0; i < 8; ++i) { a2[kt][i] = lo[i]; a2[kt][i + 8] = hi[i]; }
        }

        // transform GEMM (bias in C) + ssp -> out
#pragma unroll
        for (int nt = 0; nt < 4; ++nt) {
            v8f c = wmma16(a2[0], bWt[0][nt], cbt[nt]);
            c = wmma16(a2[1], bWt[1][nt], c);
            const int f = nt * 16 + n;
#pragma unroll
            for (int r = 0; r < 8; ++r) {
                const int a = a0 + r + 8 * half;
                if (a < N) out[a * FDIM + f] = ssp_f(c[r]);
            }
        }
    }
}

// ---------------------------------------------------------------------------
extern "C" void kernel_launch(void* const* d_in, const int* in_sizes, int n_in,
                              void* d_out, int out_size, void* d_ws, size_t ws_size,
                              hipStream_t stream) {
    // setup_inputs order:
    // 0:q 1:mu_field 2:f_ij 3:d_ij 4:v_ij 5:idx_i 6:idx_j 7:rcut_ij
    // 8:W1 9:b1 10:W2 11:b2 12:Wt 13:bt
    const float* mu_field = (const float*)d_in[1];
    const float* f_ij     = (const float*)d_in[2];
    const float* d_ijp    = (const float*)d_in[3];
    const float* v_ij     = (const float*)d_in[4];
    const int*   idx_i    = (const int*)d_in[5];
    const int*   idx_j    = (const int*)d_in[6];
    const float* rcut_ij  = (const float*)d_in[7];
    const float* W1       = (const float*)d_in[8];
    const float* b1       = (const float*)d_in[9];
    const float* W2       = (const float*)d_in[10];
    const float* b2       = (const float*)d_in[11];
    const float* Wt       = (const float*)d_in[12];
    const float* bt       = (const float*)d_in[13];
    float* out      = (float*)d_out;
    float* tensor_i = (float*)d_ws;              // [N, 3, 64] f32 accumulator

    const int P = in_sizes[3];                   // n_pairs
    const int N = in_sizes[1] / (3 * FDIM);      // n_atoms

    // 1) zero the scatter accumulator (harness does not re-zero d_ws)
    const int zc = N * 3 * FDIM;
    zero_kernel<<<(zc + 255) / 256, 256, 0, stream>>>(tensor_i, zc);

    // 2) pair path: persistent waves amortize weight conversion
    const int tilesA  = (P + 15) / 16;
    int blocksA = (tilesA + WPB - 1) / WPB;
    if (blocksA > 512) blocksA = 512;            // 4096 persistent waves
    pair_kernel<<<blocksA, 32 * WPB, 0, stream>>>(
        mu_field, f_ij, d_ijp, v_ij, idx_i, idx_j, rcut_ij,
        W1, b1, W2, b2, tensor_i, P, tilesA);

    // 3) atom path
    const int tilesB  = (N + 15) / 16;
    int blocksB = (tilesB + WPB - 1) / WPB;
    if (blocksB > 128) blocksB = 128;            // 1024 persistent waves
    atom_kernel<<<blocksB, 32 * WPB, 0, stream>>>(
        mu_field, tensor_i, Wt, bt, out, N, tilesB);
}
